// MixtureOfExperts_67164698575443
// MI455X (gfx1250) — matrix operand order repaired and checked by reading, not compiled, not run on hardware
//
#include <hip/hip_runtime.h>
#include <hip/hip_bf16.h>

// MoE: B=2,S=1024,D=1024,H=4096,E=8,K=2. Top-2 routed compute only:
// 68.7 GFLOP, ~330MB HBM traffic -> memory-bound near 14us @ 23.3TB/s.
// bf16 WMMA (f32 accum) GEMMs, 128x64 block tiles, 64-wide k slabs,
// packed v_cvt_pk_bf16_f32 conversion, async global->LDS for bf16 A in fc2.

#define Dz 1024
#define Hz 4096
#define Ez 8
#define NTOK 2048

typedef __attribute__((ext_vector_type(16))) __bf16 bf16x16;
typedef __attribute__((ext_vector_type(8)))  __bf16 bf16x8;
typedef __attribute__((ext_vector_type(8)))  float  f32x8;

// packed f32x2 -> bf16x2 (single v_cvt_pk_bf16_f32 when available)
__device__ __forceinline__ unsigned pk2(float a, float b) {
#if __has_builtin(__builtin_amdgcn_cvt_pk_bf16_f32)
  return __builtin_bit_cast(unsigned, __builtin_amdgcn_cvt_pk_bf16_f32(a, b));
#else
  union { __bf16 h[2]; unsigned u; } o;
  o.h[0] = (__bf16)a; o.h[1] = (__bf16)b; return o.u;
#endif
}

// 8 floats -> 8 bf16 packed in a uint4 (4 cvt_pk + single b128 store at caller)
__device__ __forceinline__ uint4 cvt8(float4 u, float4 v) {
  uint4 r;
  r.x = pk2(u.x, u.y); r.y = pk2(u.z, u.w);
  r.z = pk2(v.x, v.y); r.w = pk2(v.z, v.w);
  return r;
}

// ---------------- Router: logits, top-2 softmax, counting-sort, aux sums ----
__global__ __launch_bounds__(256) void moe_router(
    const float* __restrict__ x, const float* __restrict__ Wg,
    int* __restrict__ cnt, int* __restrict__ tok_list,
    float* __restrict__ prob_list, float* __restrict__ sum_probs)
{
  __shared__ float sWg[Ez * Dz];                    // 32 KB
  int tid = threadIdx.x;
  for (int i = tid; i < Ez * Dz; i += 256) sWg[i] = Wg[i];
  __syncthreads();

  int wave = tid >> 5, lane = tid & 31;
  int tok0 = (blockIdx.x * 8 + wave) * 32;
  for (int t = 0; t < 32; ++t) {
    int token = tok0 + t;
    float acc[Ez];
    #pragma unroll
    for (int e = 0; e < Ez; ++e) acc[e] = 0.f;
    const float* xp = x + (size_t)token * Dz;
    for (int d = lane; d < Dz; d += 32) {
      float xv = xp[d];
      #pragma unroll
      for (int e = 0; e < Ez; ++e) acc[e] = fmaf(xv, sWg[e * Dz + d], acc[e]);
    }
    #pragma unroll
    for (int e = 0; e < Ez; ++e)
      for (int off = 16; off > 0; off >>= 1) acc[e] += __shfl_down(acc[e], off);

    if (lane == 0) {
      int i0 = 0;
      #pragma unroll
      for (int e = 1; e < Ez; ++e) if (acc[e] > acc[i0]) i0 = e;
      int i1 = (i0 == 0) ? 1 : 0;
      #pragma unroll
      for (int e = 0; e < Ez; ++e) if (e != i0 && acc[e] > acc[i1]) i1 = e;
      float m = acc[i0], se = 0.f, pr[Ez];
      #pragma unroll
      for (int e = 0; e < Ez; ++e) { pr[e] = __expf(acc[e] - m); se += pr[e]; }
      float inv = 1.f / se;
      #pragma unroll
      for (int e = 0; e < Ez; ++e) atomicAdd(&sum_probs[e], pr[e] * inv);
      float p0 = 1.f / (1.f + __expf(acc[i1] - acc[i0]));
      float p1 = 1.f - p0;
      int s0 = atomicAdd(&cnt[i0], 1);
      tok_list[i0 * NTOK + s0] = token; prob_list[i0 * NTOK + s0] = p0;
      int s1 = atomicAdd(&cnt[i1], 1);
      tok_list[i1 * NTOK + s1] = token; prob_list[i1 * NTOK + s1] = p1;
    }
  }
}

// ---------------- Offsets (compact h rows) + aux loss ----------------------
__global__ void moe_aux(const int* __restrict__ cnt,
                        const float* __restrict__ sum_probs,
                        int* __restrict__ offsets, float* __restrict__ aux_out)
{
  if (threadIdx.x == 0) {
    int run = 0; float s = 0.f;
    for (int e = 0; e < Ez; ++e) {
      offsets[e] = run; run += cnt[e];
      s += sum_probs[e] * (float)cnt[e];
    }
    aux_out[0] = (float)Ez * s / ((float)NTOK * (float)NTOK);
  }
}

// Per-wave compute of a 32x32 sub-tile over one 64-wide k slab (2 k-steps,
// 8 WMMAs). At rows = block M, Bt rows = block N, both [.][64] bf16.
#define WMMA_SLAB(AT, BT, wmv, wnv)                                            \
  _Pragma("unroll")                                                            \
  for (int ks = 0; ks < 64; ks += 32) {                                        \
    bf16x16 a0, a1, bv0, bv1;                                                  \
    {                                                                          \
      bf16x8 lo = *(const bf16x8*)&AT[(wmv) * 32 + ln][ks + hh * 16];          \
      bf16x8 hi = *(const bf16x8*)&AT[(wmv) * 32 + ln][ks + hh * 16 + 8];      \
      _Pragma("unroll")                                                        \
      for (int i = 0; i < 8; ++i) { a0[i] = lo[i]; a0[8 + i] = hi[i]; }        \
    }                                                                          \
    {                                                                          \
      bf16x8 lo = *(const bf16x8*)&AT[(wmv) * 32 + 16 + ln][ks + hh * 16];     \
      bf16x8 hi = *(const bf16x8*)&AT[(wmv) * 32 + 16 + ln][ks + hh * 16 + 8]; \
      _Pragma("unroll")                                                        \
      for (int i = 0; i < 8; ++i) { a1[i] = lo[i]; a1[8 + i] = hi[i]; }        \
    }                                                                          \
    bv0 = *(const bf16x16*)&BT[(wnv) * 32 + ln][ks + hh * 16];                 \
    bv1 = *(const bf16x16*)&BT[(wnv) * 32 + 16 + ln][ks + hh * 16];            \
    acc00 = __builtin_amdgcn_wmma_f32_16x16x32_bf16(false, a0, false, bv0, (short)0, acc00, false, false); \
    acc01 = __builtin_amdgcn_wmma_f32_16x16x32_bf16(false, a0, false, bv1, (short)0, acc01, false, false); \
    acc10 = __builtin_amdgcn_wmma_f32_16x16x32_bf16(false, a1, false, bv0, (short)0, acc10, false, false); \
    acc11 = __builtin_amdgcn_wmma_f32_16x16x32_bf16(false, a1, false, bv1, (short)0, acc11, false, false); \
  }

// ---------------- fc1: h = relu(x_gathered @ W1[e]^T + b1[e]) (bf16 out) ----
// block tile: 128 tokens x 64 H, 8 waves in 4x2, each wave 32x32.
__global__ __launch_bounds__(256) void moe_fc1(
    const float* __restrict__ x, const float* __restrict__ W1,
    const float* __restrict__ b1, const int* __restrict__ cnt,
    const int* __restrict__ offsets, const int* __restrict__ tok_list,
    __bf16* __restrict__ hbuf)
{
  const int NT = Hz / 64;             // 64 n-tiles
  const int MT = NTOK / 128;          // 16 m-tiles (worst case)
  int bx = blockIdx.x;
  int e  = bx / (MT * NT);
  int r  = bx % (MT * NT);
  int mt = r / NT;
  int nt = r % NT;
  int cnte = cnt[e];
  if (mt * 128 >= cnte) return;

  __shared__ __bf16 At[128][64];      // 16 KB
  __shared__ __bf16 Bt[64][64];       // 8 KB
  __shared__ int rowtok[128];

  int tid = threadIdx.x;
  if (tid < 128) {
    int rr = mt * 128 + tid;
    rowtok[tid] = tok_list[e * NTOK + ((rr < cnte) ? rr : 0)];
  }
  __syncthreads();

  int lane = tid & 31, wave = tid >> 5;
  int wm = wave >> 1, wn = wave & 1;         // 4x2 wave grid over 128x64 tile
  int hh = lane >> 4, ln = lane & 15;

  int arow = tid >> 1;                        // A staging: 2 threads/row, 32 floats
  int acs  = (tid & 1) * 32;
  int brow = tid >> 2;                        // B staging: 4 threads/row, 16 floats
  int bcs  = (tid & 3) * 16;
  const size_t xbase  = (size_t)rowtok[arow] * Dz;
  const size_t w1base = ((size_t)e * Hz + (size_t)nt * 64 + brow) * Dz;

  f32x8 acc00 = {}, acc01 = {}, acc10 = {}, acc11 = {};

  for (int kk = 0; kk < Dz; kk += 64) {
    {  // stage A (gathered activations, f32 -> bf16, packed cvt, b128 stores)
      const float4* p = (const float4*)(x + xbase + kk + acs);
      uint4* d = (uint4*)&At[arow][acs];
      d[0] = cvt8(p[0], p[1]);
      d[1] = cvt8(p[2], p[3]);
      d[2] = cvt8(p[4], p[5]);
      d[3] = cvt8(p[6], p[7]);
    }
    {  // stage B (W1 rows, f32 -> bf16); prefetch next k-slab of weights
      const float4* p = (const float4*)(W1 + w1base + kk + bcs);
      if (kk + 64 < Dz) __builtin_prefetch((const void*)(W1 + w1base + kk + 64 + bcs), 0, 1);
      uint4* d = (uint4*)&Bt[brow][bcs];
      d[0] = cvt8(p[0], p[1]);
      d[1] = cvt8(p[2], p[3]);
    }
    __syncthreads();
    WMMA_SLAB(At, Bt, wm, wn)
    __syncthreads();
  }

  int rowbase = offsets[e];
  #pragma unroll
  for (int jm = 0; jm < 2; ++jm) {
    #pragma unroll
    for (int jn = 0; jn < 2; ++jn) {
      const f32x8& acc = jm ? (jn ? acc11 : acc10) : (jn ? acc01 : acc00);
      int col = nt * 64 + wn * 32 + jn * 16 + ln;
      float bb = b1[e * Hz + col];
      #pragma unroll
      for (int v = 0; v < 8; ++v) {
        int gr = mt * 128 + wm * 32 + jm * 16 + v + 8 * hh;
        if (gr < cnte) {
          float val = acc[v] + bb;
          val = val > 0.f ? val : 0.f;
          hbuf[(size_t)(rowbase + gr) * Hz + col] = (__bf16)val;
        }
      }
    }
  }
}

// ---------------- fc2: out += p * (h @ W2[e]^T + b2[e]) --------------------
__global__ __launch_bounds__(256) void moe_fc2(
    const __bf16* __restrict__ hbuf, const float* __restrict__ W2,
    const float* __restrict__ b2, const int* __restrict__ cnt,
    const int* __restrict__ offsets, const int* __restrict__ tok_list,
    const float* __restrict__ prob_list, float* __restrict__ out)
{
  const int NT = Dz / 64;             // 16 n-tiles
  const int MT = NTOK / 128;          // 16 m-tiles
  int bx = blockIdx.x;
  int e  = bx / (MT * NT);
  int r  = bx % (MT * NT);
  int mt = r / NT;
  int nt = r % NT;
  int cnte = cnt[e];
  if (mt * 128 >= cnte) return;

  __shared__ __bf16 At[128][64];      // 16 KB
  __shared__ __bf16 Bt[64][64];       // 8 KB
  __shared__ int   rowtok[128];
  __shared__ float rowp[128];

  int tid = threadIdx.x;
  int rowbase = offsets[e];
  if (tid < 128) {
    int rr = mt * 128 + tid;
    int cl = (rr < cnte) ? rr : 0;
    rowtok[tid] = tok_list[e * NTOK + cl];
    rowp[tid]   = prob_list[e * NTOK + cl];
  }
  __syncthreads();

  int lane = tid & 31, wave = tid >> 5;
  int wm = wave >> 1, wn = wave & 1;
  int hh = lane >> 4, ln = lane & 15;

  int arow = tid >> 1;                // A staging: 2 threads/row, 32 halfs (64B)
  int acs  = (tid & 1) * 32;
  int brow = tid >> 2;                // B staging: 4 threads/row, 16 floats
  int bcs  = (tid & 3) * 16;
  int crow = mt * 128 + arow;
  const size_t hbase  = (size_t)(rowbase + ((crow < cnte) ? crow : 0)) * Hz;
  const size_t w2base = ((size_t)e * Dz + (size_t)nt * 64 + brow) * Hz;
  unsigned ldsA = (unsigned)(size_t)&At[arow][acs];   // LDS byte address (low 32b)

  f32x8 acc00 = {}, acc01 = {}, acc10 = {}, acc11 = {};

  for (int kk = 0; kk < Hz; kk += 64) {
    // stage A: h already bf16 -> async global->LDS copy (4x b128 per thread)
    {
      const __bf16* gsrc = hbuf + hbase + kk + acs;
      asm volatile("global_load_async_to_lds_b128 %0, %1, off"
                   :: "v"(ldsA), "v"(gsrc) : "memory");
      asm volatile("global_load_async_to_lds_b128 %0, %1, off offset:16"
                   :: "v"(ldsA), "v"(gsrc) : "memory");
      asm volatile("global_load_async_to_lds_b128 %0, %1, off offset:32"
                   :: "v"(ldsA), "v"(gsrc) : "memory");
      asm volatile("global_load_async_to_lds_b128 %0, %1, off offset:48"
                   :: "v"(ldsA), "v"(gsrc) : "memory");
    }
    {  // stage B (W2 rows, f32 -> bf16)
      const float4* p = (const float4*)(W2 + w2base + kk + bcs);
      if (kk + 64 < Hz) __builtin_prefetch((const void*)(W2 + w2base + kk + 64 + bcs), 0, 1);
      uint4* d = (uint4*)&Bt[brow][bcs];
      d[0] = cvt8(p[0], p[1]);
      d[1] = cvt8(p[2], p[3]);
    }
    asm volatile("s_wait_asynccnt 0x0" ::: "memory");
    __syncthreads();
    WMMA_SLAB(At, Bt, wm, wn)
    __syncthreads();
  }

  #pragma unroll
  for (int jm = 0; jm < 2; ++jm) {
    #pragma unroll
    for (int jn = 0; jn < 2; ++jn) {
      const f32x8& acc = jm ? (jn ? acc11 : acc10) : (jn ? acc01 : acc00);
      int col = nt * 64 + wn * 32 + jn * 16 + ln;
      float bb = b2[e * Dz + col];
      #pragma unroll
      for (int v = 0; v < 8; ++v) {
        int mloc = wm * 32 + jm * 16 + v + 8 * hh;
        int gr = mt * 128 + mloc;
        if (gr < cnte) {
          float val = rowp[mloc] * (acc[v] + bb);
          atomicAdd(&out[(size_t)rowtok[mloc] * Dz + col], val);
        }
      }
    }
  }
}

// ---------------- host launch ----------------------------------------------
extern "C" void kernel_launch(void* const* d_in, const int* in_sizes, int n_in,
                              void* d_out, int out_size, void* d_ws, size_t ws_size,
                              hipStream_t stream) {
  const float* x  = (const float*)d_in[0];
  const float* Wg = (const float*)d_in[1];
  const float* W1 = (const float*)d_in[2];
  const float* b1 = (const float*)d_in[3];
  const float* W2 = (const float*)d_in[4];
  const float* b2 = (const float*)d_in[5];
  float* out = (float*)d_out;

  char* ws = (char*)d_ws;
  int*    cnt       = (int*)(ws + 0);
  int*    offsets   = (int*)(ws + 64);
  float*  sum_probs = (float*)(ws + 128);
  int*    tok_list  = (int*)(ws + 256);
  float*  prob_list = (float*)(ws + 256 + Ez * NTOK * 4);
  __bf16* hbuf      = (__bf16*)(ws + 262144);   // 2*NTOK rows x Hz bf16 = 32MB

  hipMemsetAsync(ws, 0, 256, stream);                                   // cnt + sums
  hipMemsetAsync(out, 0, (size_t)NTOK * Dz * sizeof(float), stream);    // accum target

  moe_router<<<NTOK / 256, 256, 0, stream>>>(x, Wg, cnt, tok_list, prob_list, sum_probs);
  moe_aux<<<1, 32, 0, stream>>>(cnt, sum_probs, offsets, out + (size_t)NTOK * Dz);
  moe_fc1<<<Ez * (NTOK / 128) * (Hz / 64), 256, 0, stream>>>(x, W1, b1, cnt, offsets, tok_list, hbuf);
  moe_fc2<<<Ez * (NTOK / 128) * (Dz / 64), 256, 0, stream>>>(hbuf, W2, b2, cnt, offsets, tok_list, prob_list, out);
}
